// DenseGATConv_28501402976490
// MI455X (gfx1250) — matrix operand order repaired and checked by reading, not compiled, not run on hardware
//
#include <hip/hip_runtime.h>
#include <hip/hip_bf16.h>

#define N_NODES 4096
#define IN_C    256
#define HEADS   4
#define OUT_C   64
#define HC      256   /* HEADS*OUT_C */
#define NEG_SLOPE 0.2f
#define JSPLIT  8
#define JCHUNK  (N_NODES / JSPLIT)   /* 512 j per block */

typedef __attribute__((ext_vector_type(2)))  float   v2f;
typedef __attribute__((ext_vector_type(8)))  float   v8f;
typedef __attribute__((ext_vector_type(16))) __bf16  v16bf;

__device__ __forceinline__ float lrelu(float z) { return z >= 0.f ? z : NEG_SLOPE * z; }

// ---------------------------------------------------------------------------
// K1: h[N,HC] = x[N,IN_C] @ W[IN_C,HC]   (fp32 WMMA 16x16x4)
//     also writes hT[HC,N] in bf16 (transposed, for contiguous-K B loads in K3)
// ---------------------------------------------------------------------------
__global__ __launch_bounds__(32) void k_gemm_h(const float* __restrict__ x,
                                               const float* __restrict__ W,
                                               float* __restrict__ h,
                                               __bf16* __restrict__ hT) {
    const int lane = threadIdx.x;
    const int col0 = (blockIdx.x & 15) * 16;
    const int row0 = (blockIdx.x >> 4) * 16;
    const int m    = lane & 15;
    const int hi   = lane >> 4;

    v8f acc = {};
    const float* xrow = x + (size_t)(row0 + m) * IN_C;
    for (int k = 0; k < IN_C; k += 4) {
        v2f a = *(const v2f*)(xrow + k + 2 * hi);            // A 16x4 f32
        v2f b;                                               // B 4x16 f32
        b.x = W[(size_t)(k + 2 * hi    ) * HC + col0 + m];
        b.y = W[(size_t)(k + 2 * hi + 1) * HC + col0 + m];
        acc = __builtin_amdgcn_wmma_f32_16x16x4_f32(false, a, false, b,
                                                    (short)0, acc, false, false);
    }
#pragma unroll
    for (int v = 0; v < 8; ++v) {
        const int row = row0 + v + 8 * hi;
        const int col = col0 + m;
        const float val = acc[v];
        h [(size_t)row * HC + col]      = val;
        hT[(size_t)col * N_NODES + row] = (__bf16)val;
    }
}

// ---------------------------------------------------------------------------
// K2a: el[n,h] = sum_c h[n,h,c]*att_l[h,c] ; er likewise (one block per row)
// ---------------------------------------------------------------------------
__global__ __launch_bounds__(256) void k_attn_coef(const float* __restrict__ h,
                                                   const float* __restrict__ att_l,
                                                   const float* __restrict__ att_r,
                                                   float* __restrict__ el,
                                                   float* __restrict__ er) {
    __shared__ float sl[256];
    __shared__ float sr[256];
    const int n = blockIdx.x, t = threadIdx.x;
    const float v = h[(size_t)n * HC + t];
    sl[t] = v * att_l[t];
    sr[t] = v * att_r[t];
    __syncthreads();
    for (int s = 32; s > 0; s >>= 1) {
        if ((t & 63) < s) { sl[t] += sl[t + s]; sr[t] += sr[t + s]; }
        __syncthreads();
    }
    if ((t & 63) == 0) {
        const int hd = t >> 6;
        el[n * HEADS + hd] = sl[t];
        er[n * HEADS + hd] = sr[t];
    }
}

// ---------------------------------------------------------------------------
// K2b: per (row i, head): masked max of er (lrelu is monotone) and exp-sum.
// Second pass re-reads the same 16KB adj row -> L0/L2 hit, no extra HBM.
// ---------------------------------------------------------------------------
__global__ __launch_bounds__(256) void k_row_stats(const int* __restrict__ adj,
                                                   const float* __restrict__ el,
                                                   const float* __restrict__ er,
                                                   float* __restrict__ emax,
                                                   float* __restrict__ ssum) {
    __shared__ float red[256];
    const int i = blockIdx.x, t = threadIdx.x;
    const int* arow = adj + (size_t)i * N_NODES;
    const float4* er4 = (const float4*)er;
    const float NINF = -__builtin_inff();

    float mx[HEADS] = {NINF, NINF, NINF, NINF};
    for (int j = t; j < N_NODES; j += 256) {
        if (arow[j] > 0) {
            const float4 e = er4[j];
            mx[0] = fmaxf(mx[0], e.x); mx[1] = fmaxf(mx[1], e.y);
            mx[2] = fmaxf(mx[2], e.z); mx[3] = fmaxf(mx[3], e.w);
        }
    }
    float Mh[HEADS];
#pragma unroll
    for (int hd = 0; hd < HEADS; ++hd) {
        red[t] = mx[hd]; __syncthreads();
        for (int s = 128; s > 0; s >>= 1) {
            if (t < s) red[t] = fmaxf(red[t], red[t + s]);
            __syncthreads();
        }
        Mh[hd] = red[0]; __syncthreads();
    }
    float elv[HEADS], emx[HEADS];
#pragma unroll
    for (int hd = 0; hd < HEADS; ++hd) {
        elv[hd] = el[i * HEADS + hd];
        emx[hd] = (Mh[hd] == NINF) ? 0.f : lrelu(elv[hd] + Mh[hd]);
    }
    float sm[HEADS] = {0.f, 0.f, 0.f, 0.f};
    for (int j = t; j < N_NODES; j += 256) {
        if (arow[j] > 0) {
            const float4 e = er4[j];
            sm[0] += __expf(lrelu(elv[0] + e.x) - emx[0]);
            sm[1] += __expf(lrelu(elv[1] + e.y) - emx[1]);
            sm[2] += __expf(lrelu(elv[2] + e.z) - emx[2]);
            sm[3] += __expf(lrelu(elv[3] + e.w) - emx[3]);
        }
    }
#pragma unroll
    for (int hd = 0; hd < HEADS; ++hd) {
        red[t] = sm[hd]; __syncthreads();
        for (int s = 128; s > 0; s >>= 1) {
            if (t < s) red[t] += red[t + s];
            __syncthreads();
        }
        if (t == 0) {
            ssum[i * HEADS + hd] = red[0];
            emax[i * HEADS + hd] = emx[hd];
        }
        __syncthreads();
    }
}

// ---------------------------------------------------------------------------
// K-init: out = bias (broadcast), so K3 can atomically accumulate into it.
// ---------------------------------------------------------------------------
__global__ __launch_bounds__(256) void k_init_out(const float* __restrict__ bias,
                                                  float* __restrict__ out) {
    out[(size_t)blockIdx.x * HC + threadIdx.x] = bias[threadIdx.x];
}

// ---------------------------------------------------------------------------
// K3: out[i,hd,c] += (1/s_i) * sum_{j in chunk} alpha'_{ij} * h[j,hd,c]
// 16 rows x 512 j per block; 4 waves = 4 heads; bf16 WMMA 16x16x32.
// Ping-pong LDS staging of adj/er tiles to overlap global fetch with WMMA.
// ---------------------------------------------------------------------------
__global__ __launch_bounds__(128) void k_aggregate(const int* __restrict__ adj,
                                                   const float* __restrict__ el,
                                                   const float* __restrict__ er,
                                                   const float* __restrict__ emax,
                                                   const float* __restrict__ ssum,
                                                   const __bf16* __restrict__ hT,
                                                   float* __restrict__ out) {
    __shared__ int   adjT[2][16][32];
    __shared__ float erT[2][32][HEADS];

    const int t     = threadIdx.x;
    const int hd    = t >> 5;          // wave -> head
    const int lane  = t & 31;
    const int row0  = blockIdx.x * 16;
    const int jbeg  = blockIdx.y * JCHUNK;
    const int mA    = lane & 15;       // A-row (and D-column) role
    const int khalf = lane >> 4;

    const float el_m  = el  [(row0 + mA) * HEADS + hd];
    const float emx_m = emax[(row0 + mA) * HEADS + hd];

    // staging registers for the software pipeline
    int   aReg[4];
    float eReg;

    auto fetch = [&](int j0) {
#pragma unroll
        for (int r = 0; r < 4; ++r) {
            const int e = t + 128 * r;
            aReg[r] = adj[(size_t)(row0 + (e >> 5)) * N_NODES + j0 + (e & 31)];
        }
        eReg = er[(size_t)(j0 + (t >> 2)) * HEADS + (t & 3)];
    };
    auto commit = [&](int buf) {
#pragma unroll
        for (int r = 0; r < 4; ++r) {
            const int e = t + 128 * r;
            adjT[buf][e >> 5][e & 31] = aReg[r];
        }
        erT[buf][t >> 2][t & 3] = eReg;
    };

    v8f acc0 = {}, acc1 = {}, acc2 = {}, acc3 = {};

    fetch(jbeg);
    commit(0);
    __syncthreads();

    const int NITER = JCHUNK / 32;   // 16
    for (int it = 0; it < NITER; ++it) {
        const int j0  = jbeg + it * 32;
        const int buf = it & 1;

        if (it + 1 < NITER) fetch(j0 + 32);   // hide global latency behind WMMA

        // A fragment 16x32 bf16: element e -> K = e + 16*khalf
        v16bf afrag;
#pragma unroll
        for (int e = 0; e < 16; ++e) {
            const int k = e + 16 * khalf;
            float a = 0.f;
            if (adjT[buf][mA][k] > 0)
                a = __expf(lrelu(el_m + erT[buf][k][hd]) - emx_m);
            afrag[e] = (__bf16)a;
        }

        // B fragments 32x16 bf16: per-lane column, K contiguous in hT (L2-hot)
        const __bf16* bb =
            hT + (size_t)(hd * OUT_C + mA) * N_NODES + j0 + 16 * khalf;
        const v16bf b0 = *(const v16bf*)(bb + (size_t) 0 * N_NODES);
        const v16bf b1 = *(const v16bf*)(bb + (size_t)16 * N_NODES);
        const v16bf b2 = *(const v16bf*)(bb + (size_t)32 * N_NODES);
        const v16bf b3 = *(const v16bf*)(bb + (size_t)48 * N_NODES);

        acc0 = __builtin_amdgcn_wmma_f32_16x16x32_bf16(false, afrag, false, b0,
                                                       (short)0, acc0, false, false);
        acc1 = __builtin_amdgcn_wmma_f32_16x16x32_bf16(false, afrag, false, b1,
                                                       (short)0, acc1, false, false);
        acc2 = __builtin_amdgcn_wmma_f32_16x16x32_bf16(false, afrag, false, b2,
                                                       (short)0, acc2, false, false);
        acc3 = __builtin_amdgcn_wmma_f32_16x16x32_bf16(false, afrag, false, b3,
                                                       (short)0, acc3, false, false);

        if (it + 1 < NITER) commit((it + 1) & 1);
        __syncthreads();
    }

    // D layout: element v -> row = row0 + v + 8*khalf, col = mA within tile.
    // 1/s is uniform per (row, head) so it commutes with the j-split.
#pragma unroll
    for (int v = 0; v < 8; ++v) {
        const int row = row0 + v + 8 * khalf;
        const float s = ssum[row * HEADS + hd];
        const float inv = (s > 0.f) ? (1.f / s) : 0.f;
        float* orow = out + (size_t)row * HC + hd * OUT_C + mA;
        __hip_atomic_fetch_add(&orow[ 0], acc0[v] * inv,
                               __ATOMIC_RELAXED, __HIP_MEMORY_SCOPE_AGENT);
        __hip_atomic_fetch_add(&orow[16], acc1[v] * inv,
                               __ATOMIC_RELAXED, __HIP_MEMORY_SCOPE_AGENT);
        __hip_atomic_fetch_add(&orow[32], acc2[v] * inv,
                               __ATOMIC_RELAXED, __HIP_MEMORY_SCOPE_AGENT);
        __hip_atomic_fetch_add(&orow[48], acc3[v] * inv,
                               __ATOMIC_RELAXED, __HIP_MEMORY_SCOPE_AGENT);
    }
}

// ---------------------------------------------------------------------------
extern "C" void kernel_launch(void* const* d_in, const int* in_sizes, int n_in,
                              void* d_out, int out_size, void* d_ws, size_t ws_size,
                              hipStream_t stream) {
    const float* x     = (const float*)d_in[0];
    const int*   adj   = (const int*)  d_in[1];
    const float* W     = (const float*)d_in[2];
    const float* att_l = (const float*)d_in[3];
    const float* att_r = (const float*)d_in[4];
    const float* bias  = (const float*)d_in[5];
    float* out = (float*)d_out;

    char* ws = (char*)d_ws;
    float*  h    = (float*) (ws);                              // 4 MB
    __bf16* hT   = (__bf16*)(ws + 4u * 1024u * 1024u);         // 2 MB
    float*  el   = (float*) (ws + 6u * 1024u * 1024u);         // 64 KB each
    float*  er   = (float*) (ws + 6u * 1024u * 1024u + 65536u);
    float*  emax = (float*) (ws + 6u * 1024u * 1024u + 131072u);
    float*  ssum = (float*) (ws + 6u * 1024u * 1024u + 196608u);

    k_gemm_h   <<<4096, 32, 0, stream>>>(x, W, h, hT);
    k_attn_coef<<<4096, 256, 0, stream>>>(h, att_l, att_r, el, er);
    k_row_stats<<<4096, 256, 0, stream>>>(adj, el, er, emax, ssum);
    k_init_out <<<4096, 256, 0, stream>>>(bias, out);
    k_aggregate<<<dim3(N_NODES / 16, JSPLIT), 128, 0, stream>>>(
        adj, el, er, emax, ssum, hT, out);
}
